// GINEStateEncoder_61555471286658
// MI455X (gfx1250) — matrix operand order repaired and checked by reading, not compiled, not run on hardware
//
#include <hip/hip_runtime.h>

// ---------------------------------------------------------------------------
// CDNA5 (gfx1250) GINE encoder: fused edge-linear (WMMA bf16) + gather/ReLU/
// atomic-scatter, WMMA node MLP + BN + ReLU, segment-mean pool.
// wave32; v_wmma_f32_16x16x32_bf16 (A 16x32, B 32x16, C/D 16x16 f32).
// 32-bit byte offsets against uniform bases -> saddr global forms.
// Pairwise f32->bf16 conversion so ISel can use a packed cvt if available.
// ---------------------------------------------------------------------------

typedef __attribute__((ext_vector_type(16))) __bf16 v16bf;
typedef __attribute__((ext_vector_type(8)))  float  v8f;
typedef __attribute__((ext_vector_type(2)))  float  v2f;
typedef __attribute__((ext_vector_type(2)))  __bf16 v2bf;

__device__ __forceinline__ v2bf f2bf2(float a, float b) {
  v2f v = {a, b};
  return __builtin_convertvector(v, v2bf);   // fptrunc pair -> pk cvt if native
}

__device__ __forceinline__ __bf16 f2bf(float x) {
  return f2bf2(x, x)[0];
}

__device__ __forceinline__ float ldg_off(const float* base, unsigned byteoff) {
  return *(const float*)((const char*)base + byteoff);
}

// ---------------------------------------------------------------------------
// Edge kernel: tiles of 16 edges; e = edge_attr(16x32) @ lew(32xDIN) + leb via
// one WMMA per 16-col tile; m = relu(x[src] + e); atomicAdd into agg[dst].
// 4 waves/block, TILES 16-edge tiles per wave; B fragments held in registers.
// Gathers batched ahead of the WMMA so load latency overlaps the matrix op.
// ---------------------------------------------------------------------------
template<int DIN, int TILES>
__global__ __launch_bounds__(128)
void gine_edge_kernel(const float* __restrict__ x,      // [N, DIN]
                      const float* __restrict__ ea,     // [E, 32]
                      const int*   __restrict__ src,
                      const int*   __restrict__ dst,
                      const float* __restrict__ lew,    // [32, DIN]
                      const float* __restrict__ leb,    // [DIN]
                      float*       __restrict__ agg,    // [N, DIN]
                      int nE)
{
  __shared__ alignas(32) __bf16 lwT[DIN * 32];   // transposed: lwT[n*32 + k]
  __shared__ float lbias[DIN];

  const int tid = threadIdx.x;
  // Stage transposed bf16 weights; pairs of consecutive k share one store.
  for (int i = tid * 2; i < DIN * 32; i += 256) {
    int n = i >> 5, k = i & 31;
    v2bf p = f2bf2(lew[k * DIN + n], lew[(k + 1) * DIN + n]);
    *(v2bf*)&lwT[i] = p;
  }
  for (int i = tid; i < DIN; i += 128) lbias[i] = leb[i];
  __syncthreads();

  const int wave = tid >> 5;
  const int lane = tid & 31;
  const int ln   = lane & 15;   // A row m / C column n
  const int hi   = lane >> 4;   // lane-group

  // Wave-invariant B fragments (weights) + biases -> registers.
  v16bf bfr[DIN / 16];
  float bias[DIN / 16];
#pragma unroll
  for (int nt = 0; nt < DIN / 16; ++nt) {
    int col = nt * 16 + ln;
    bfr[nt]  = *(const v16bf*)&lwT[col * 32 + hi * 16];
    bias[nt] = lbias[col];
  }

  const int eb0 = (blockIdx.x * 4 + wave) * (16 * TILES);
#pragma unroll 1
  for (int t = 0; t < TILES; ++t) {
    const int eb = eb0 + 16 * t;
    if (eb >= nE) break;

    // A fragment: 16x32 edge_attr tile (contiguous k = 16*hi + s, as in B), pairwise cvt.
    const float* rowp = ea + (size_t)(eb + ln) * 32;
    if (t + 1 < TILES)                       // next tile's rows -> prefetch
      __builtin_prefetch(rowp + 16 * 32, 0, 3);
    v16bf a;
#pragma unroll
    for (int j = 0; j < 8; j += 2) {
      v2bf p0 = f2bf2(rowp[hi * 16 + j],     rowp[hi * 16 + j + 1]);
      v2bf p1 = f2bf2(rowp[hi * 16 + 8 + j], rowp[hi * 16 + 8 + j + 1]);
      a[j]     = p0[0]; a[j + 1]     = p0[1];
      a[j + 8] = p1[0]; a[j + 8 + 1] = p1[1];
    }

    // The 8 C rows this lane owns: m = i + 8*hi; 32-bit row byte offsets.
    unsigned sb[8], db[8];
#pragma unroll
    for (int i = 0; i < 8; ++i) {
      int e = eb + i + 8 * hi;
      sb[i] = (unsigned)src[e] * (unsigned)(DIN * 4);
      db[i] = (unsigned)dst[e] * (unsigned)(DIN * 4);
    }

#pragma unroll
    for (int nt = 0; nt < DIN / 16; ++nt) {
      const unsigned cb = (unsigned)((nt * 16 + ln) << 2);

      // Launch all 8 independent gathers first ...
      float xv[8];
#pragma unroll
      for (int i = 0; i < 8; ++i) xv[i] = ldg_off(x, sb[i] + cb);

      // ... overlap their latency with the WMMA ...
      v8f c;
#pragma unroll
      for (int i = 0; i < 8; ++i) c[i] = bias[nt];
      c = __builtin_amdgcn_wmma_f32_16x16x32_bf16(false, a, false, bfr[nt],
                                                  (short)0, c, false, false);

      // ... then relu + scatter.
#pragma unroll
      for (int i = 0; i < 8; ++i) {
        float v = c[i] + xv[i];
        v = v > 0.f ? v : 0.f;
        atomicAdd((float*)((char*)agg + (db[i] + cb)), v);
      }
    }
  }
}

// ---------------------------------------------------------------------------
// Node kernel: h = x + agg; t = relu(h @ w1 + b1); o = t @ w2 + b2;
// o = relu((o - mu) * g * rsqrt(var+eps) + beta). 16 nodes/tile, TILES/wave.
// ---------------------------------------------------------------------------
template<int DIN, int DOUT, int TILES>
__global__ __launch_bounds__(128)
void gine_node_kernel(const float* __restrict__ xin,  // [N, DIN]
                      const float* __restrict__ agg,  // [N, DIN]
                      const float* __restrict__ w1,   // [DIN, DOUT]
                      const float* __restrict__ b1,
                      const float* __restrict__ w2,   // [DOUT, DOUT]
                      const float* __restrict__ b2,
                      const float* __restrict__ bng,
                      const float* __restrict__ bnb,
                      const float* __restrict__ bnm,
                      const float* __restrict__ bnv,
                      float*       __restrict__ out,  // [N, DOUT]
                      int nN)
{
  __shared__ alignas(32) __bf16 w1T[DOUT * DIN];    // [n][k]
  __shared__ alignas(32) __bf16 w2T[DOUT * DOUT];   // [n][k]
  __shared__ alignas(32) __bf16 tbf[4 * 16 * DOUT]; // per-wave 16 x DOUT

  const int tid = threadIdx.x;
  for (int i = tid * 2; i < DOUT * DIN; i += 256) {
    int n = i / DIN, k = i % DIN;           // consecutive i -> consecutive k
    v2bf p = f2bf2(w1[k * DOUT + n], w1[(k + 1) * DOUT + n]);
    *(v2bf*)&w1T[i] = p;
  }
  for (int i = tid * 2; i < DOUT * DOUT; i += 256) {
    int n = i / DOUT, k = i % DOUT;
    v2bf p = f2bf2(w2[k * DOUT + n], w2[(k + 1) * DOUT + n]);
    *(v2bf*)&w2T[i] = p;
  }
  __syncthreads();

  const int wave = tid >> 5;
  const int lane = tid & 31;
  const int ln   = lane & 15;
  const int hi   = lane >> 4;

  // Per-column parameters, invariant across tiles -> registers.
  float bias1[DOUT / 16], bias2[DOUT / 16];
  float scale[DOUT / 16], mu[DOUT / 16], beta[DOUT / 16];
#pragma unroll
  for (int nt = 0; nt < DOUT / 16; ++nt) {
    int col = nt * 16 + ln;
    bias1[nt] = b1[col];
    bias2[nt] = b2[col];
    scale[nt] = bng[col] * rsqrtf(bnv[col] + 1e-5f);
    mu[nt]    = bnm[col];
    beta[nt]  = bnb[col];
  }

  __bf16* tw = &tbf[wave * 16 * DOUT];
  const int rb0 = (blockIdx.x * 4 + wave) * (16 * TILES);

#pragma unroll 1
  for (int t = 0; t < TILES; ++t) {
    const int rb = rb0 + 16 * t;
    if (rb >= nN) break;

    // A fragments for h = x + agg (16 x DIN), K-chunked by 32, pairwise cvt.
    v16bf afr[DIN / 32];
    {
      const unsigned ro = (unsigned)(rb + ln) * (unsigned)(DIN * 4);
#pragma unroll
      for (int kc = 0; kc < DIN / 32; ++kc) {
#pragma unroll
        for (int j = 0; j < 8; j += 2) {
          unsigned o0 = ro + (unsigned)((kc * 32 + hi * 16 + j) << 2);
          unsigned o1 = ro + (unsigned)((kc * 32 + hi * 16 + 8 + j) << 2);
          v2bf p0 = f2bf2(ldg_off(xin, o0)     + ldg_off(agg, o0),
                          ldg_off(xin, o0 + 4) + ldg_off(agg, o0 + 4));
          v2bf p1 = f2bf2(ldg_off(xin, o1)     + ldg_off(agg, o1),
                          ldg_off(xin, o1 + 4) + ldg_off(agg, o1 + 4));
          afr[kc][j]     = p0[0]; afr[kc][j + 1]     = p0[1];
          afr[kc][j + 8] = p1[0]; afr[kc][j + 8 + 1] = p1[1];
        }
      }
    }

    // GEMM1 + ReLU -> wave-private LDS tile (bf16)
#pragma unroll
    for (int nt = 0; nt < DOUT / 16; ++nt) {
      const int col = nt * 16 + ln;
      v8f c;
#pragma unroll
      for (int i = 0; i < 8; ++i) c[i] = bias1[nt];
#pragma unroll
      for (int kc = 0; kc < DIN / 32; ++kc) {
        v16bf b = *(const v16bf*)&w1T[col * DIN + kc * 32 + hi * 16];
        c = __builtin_amdgcn_wmma_f32_16x16x32_bf16(false, afr[kc], false, b,
                                                    (short)0, c, false, false);
      }
#pragma unroll
      for (int i = 0; i < 8; ++i) {
        float v = c[i];
        v = v > 0.f ? v : 0.f;
        tw[(i + 8 * hi) * DOUT + col] = f2bf(v);  // wave-internal LDS
      }
    }

    // GEMM2 + BatchNorm + ReLU -> out
#pragma unroll
    for (int nt = 0; nt < DOUT / 16; ++nt) {
      const int col = nt * 16 + ln;
      v8f c;
#pragma unroll
      for (int i = 0; i < 8; ++i) c[i] = bias2[nt];
#pragma unroll
      for (int kc = 0; kc < DOUT / 32; ++kc) {
        v16bf a2;
#pragma unroll
        for (int j = 0; j < 8; ++j) {
          a2[j]     = tw[ln * DOUT + kc * 32 + hi * 16 + j];
          a2[j + 8] = tw[ln * DOUT + kc * 32 + hi * 16 + 8 + j];
        }
        v16bf b = *(const v16bf*)&w2T[col * DOUT + kc * 32 + hi * 16];
        c = __builtin_amdgcn_wmma_f32_16x16x32_bf16(false, a2, false, b,
                                                    (short)0, c, false, false);
      }
#pragma unroll
      for (int i = 0; i < 8; ++i) {
        float v = (c[i] - mu[nt]) * scale[nt] + beta[nt];
        v = v > 0.f ? v : 0.f;
        unsigned oo = (unsigned)(rb + i + 8 * hi) * (unsigned)(DOUT * 4)
                    + (unsigned)(col << 2);
        *(float*)((char*)out + oo) = v;
      }
    }
  }
}

// ---------------------------------------------------------------------------
// Utility / pooling kernels
// ---------------------------------------------------------------------------
__global__ void zero_kernel(float* __restrict__ p, int n) {
  int i = blockIdx.x * 256 + threadIdx.x;
  if (i < n) p[i] = 0.f;
}

__global__ void pool_acc_kernel(const float* __restrict__ h,   // [N, 96]
                                const int* __restrict__ batch,
                                float* __restrict__ sums,      // [64, 96]
                                float* __restrict__ cnts,      // [64]
                                int nN) {
  int idx = blockIdx.x * 256 + threadIdx.x;
  if (idx >= nN * 96) return;
  int n = idx / 96, c = idx % 96;
  int g = batch[n];
  atomicAdd(&sums[g * 96 + c], h[idx]);
  if (c == 0) atomicAdd(&cnts[g], 1.0f);
}

__global__ void pool_div_kernel(const float* __restrict__ sums,
                                const float* __restrict__ cnts,
                                float* __restrict__ out) {
  int idx = blockIdx.x * 256 + threadIdx.x;
  if (idx >= 64 * 96) return;
  int g = idx / 96;
  float c = cnts[g];
  out[idx] = sums[idx] / fmaxf(c, 1.0f);
}

// ---------------------------------------------------------------------------
// Launcher
// ---------------------------------------------------------------------------
extern "C" void kernel_launch(void* const* d_in, const int* in_sizes, int n_in,
                              void* d_out, int out_size, void* d_ws, size_t ws_size,
                              hipStream_t stream) {
  (void)n_in; (void)out_size; (void)ws_size;

  const float* x     = (const float*)d_in[0];
  const float* ea    = (const float*)d_in[1];
  const int*   eidx  = (const int*)d_in[2];
  const int*   batch = (const int*)d_in[3];

  const int N = in_sizes[0] / 128;      // 50000
  const int E = in_sizes[2] / 2;        // 800000
  const int* src = eidx;
  const int* dst = eidx + E;

  // Per-layer params: base index 4 + 10*l, order:
  // le_w, le_b, w1, b1, w2, b2, bn_g, bn_b, bn_m, bn_v
  const float* P[3][10];
  for (int l = 0; l < 3; ++l)
    for (int j = 0; j < 10; ++j)
      P[l][j] = (const float*)d_in[4 + 10 * l + j];

  float* ws   = (float*)d_ws;
  float* agg  = ws;                           // N*128 max
  float* hA   = agg + (size_t)N * 128;        // N*96 max
  float* hB   = hA + (size_t)N * 96;          // N*96 max
  float* sums = hB + (size_t)N * 96;          // 64*96
  float* cnts = sums + 64 * 96;               // 64

  constexpr int ET = 4;  // edge tiles per wave  (64 edges/wave, 256/block)
  constexpr int NT = 4;  // node tiles per wave  (64 nodes/wave, 256/block)
  const int edgeBlocks = (E + 64 * ET - 1) / (64 * ET);
  const int nodeBlocks = (N + 64 * NT - 1) / (64 * NT);

  auto zgrid = [](int n) { return (n + 255) / 256; };

  // ---- layer 0: din=128, dout=64 ----
  zero_kernel<<<zgrid(N * 128), 256, 0, stream>>>(agg, N * 128);
  gine_edge_kernel<128, ET><<<edgeBlocks, 128, 0, stream>>>(
      x, ea, src, dst, P[0][0], P[0][1], agg, E);
  gine_node_kernel<128, 64, NT><<<nodeBlocks, 128, 0, stream>>>(
      x, agg, P[0][2], P[0][3], P[0][4], P[0][5],
      P[0][6], P[0][7], P[0][8], P[0][9], hA, N);

  // ---- layer 1: din=64, dout=64 ----
  zero_kernel<<<zgrid(N * 64), 256, 0, stream>>>(agg, N * 64);
  gine_edge_kernel<64, ET><<<edgeBlocks, 128, 0, stream>>>(
      hA, ea, src, dst, P[1][0], P[1][1], agg, E);
  gine_node_kernel<64, 64, NT><<<nodeBlocks, 128, 0, stream>>>(
      hA, agg, P[1][2], P[1][3], P[1][4], P[1][5],
      P[1][6], P[1][7], P[1][8], P[1][9], hB, N);

  // ---- layer 2: din=64, dout=96 ----
  zero_kernel<<<zgrid(N * 64), 256, 0, stream>>>(agg, N * 64);
  gine_edge_kernel<64, ET><<<edgeBlocks, 128, 0, stream>>>(
      hB, ea, src, dst, P[2][0], P[2][1], agg, E);
  gine_node_kernel<64, 96, NT><<<nodeBlocks, 128, 0, stream>>>(
      hB, agg, P[2][2], P[2][3], P[2][4], P[2][5],
      P[2][6], P[2][7], P[2][8], P[2][9], hA, N);

  // ---- global mean pool ----
  zero_kernel<<<zgrid(64 * 96 + 64), 256, 0, stream>>>(sums, 64 * 96 + 64);
  pool_acc_kernel<<<zgrid(N * 96), 256, 0, stream>>>(hA, batch, sums, cnts, N);
  pool_div_kernel<<<zgrid(64 * 96), 256, 0, stream>>>(sums, cnts, (float*)d_out);
}
